// Encoder_4836133175591
// MI455X (gfx1250) — matrix operand order repaired and checked by reading, not compiled, not run on hardware
//
#include <hip/hip_runtime.h>
#include <hip/hip_bf16.h>

#define DEVI __device__ __forceinline__

typedef _Float16 h16;
typedef __attribute__((ext_vector_type(8)))  _Float16 v8h;
typedef __attribute__((ext_vector_type(16))) _Float16 v16h;
typedef __attribute__((ext_vector_type(8)))  float    v8f;

constexpr int NN  = 1200000;   // rows
constexpr int NB  = 150000;    // segments
constexpr int NBp = 150016;    // segments padded to multiple of 64
constexpr float EPS = 1e-5f;

struct P {
  // raw inputs (f32, row-major as in JAX)
  const float *x;
  const float *kd_W1,*kd_b1,*kd_W2,*kd_b2,*kd_g1,*kd_be1;   // key_ds   9->52->96  LN
  const float *vd_W1,*vd_b1,*vd_W2,*vd_b2,*vd_g1,*vd_be1;   // val_ds   96->96->96 LN
  const float *ed_W1,*ed_b1,*ed_W2,*ed_b2;                  // enc_ds   96->96->96
  const float *km_W1,*km_b1,*km_W2,*km_b2,*km_g1,*km_be1;   // key_main 105->116->128 LN
  const float *vm_W1,*vm_b1,*vm_W2,*vm_b2,*vm_g1,*vm_be1;   // val_main 192->160->128 LN
  const float *em_W1,*em_b1,*em_W2,*em_b2;                  // enc_main 137->132->128
  const float *rank_W,*rank_b;
  // workspace
  float *mag; int *perm;
  float *y2ds;            // [NBp][96]  f32
  h16   *zdsh;            // [NBp][96]  f16
  float *y2m;             // [NBp][128] f32
  float *keytab;          // [8][96]    f32  (key_ds MLP of one_hot(0..7))
  float *em_b1p;          // [132]      f32  (enc_main b1 + W1[136,:])
  h16 *w1t_vd,*w2t_vd,*w1t_ed,*w2t_ed;            // [96][96]
  h16 *w1t_vm;            // [160][192]
  h16 *w2t_vm;            // [128][160]
  h16 *w1t_km,*w2t_km;    // [128][128] (K padded 105->128 / 116->128, N padded 116->128)
  h16 *w1t_em;            // [144][128] (N padded 132->144, K folded to 128)
  h16 *w2t_em;            // [128][160] (K padded 132->160)
  float *out;             // [NB][128]
};

// ---- WMMA fragment loaders (f16 16x16x32) -------------------------------
// A 16x32: lane r(0-15) half h: K = {k0+8h .. +7} and {k0+16+8h .. +7}
DEVI v16h ldA(const h16* buf, int stride, int row0, int k0, int lane) {
  int r = lane & 15, h = lane >> 4;
  const h16* p = buf + (size_t)(row0 + r) * stride + k0 + h * 8;
  v8h lo = *(const v8h*)p;
  v8h hi = *(const v8h*)(p + 16);
  return __builtin_shufflevector(lo, hi, 0,1,2,3,4,5,6,7,8,9,10,11,12,13,14,15);
}
// B 32x16 from transposed weights WT[n][k]: lane n(0-15) half h: K = {k0+8h .. +7} and {k0+16+8h .. +7}
DEVI v16h ldB(const h16* buf, int stride, int n0, int k0, int lane) {
  int n = lane & 15, h = lane >> 4;
  const h16* p = buf + (size_t)(n0 + n) * stride + k0 + h * 8;
  v8h lo = *(const v8h*)p;
  v8h hi = *(const v8h*)(p + 16);
  return __builtin_shufflevector(lo, hi, 0,1,2,3,4,5,6,7,8,9,10,11,12,13,14,15);
}
DEVI v8f wmmaf(v16h a, v16h b, v8f c) {
  return __builtin_amdgcn_wmma_f32_16x16x32_f16(false, a, false, b, (short)0, c, false, false);
}
DEVI v8f vzero() { v8f z = {0.f,0.f,0.f,0.f,0.f,0.f,0.f,0.f}; return z; }

// ---- prep: weight transpose/pad to f16, keytab, folded bias -------------
DEVI void convT(const float* W, h16* dst, int K, int Nn, int Kp, int Np, int t) {
  int total = Kp * Np;
  for (int e = t; e < total; e += 256) {
    int n = e / Kp, k = e % Kp;
    float v = (k < K && n < Nn) ? W[(size_t)k * Nn + n] : 0.f;
    dst[(size_t)n * Kp + k] = (h16)v;
  }
}

__global__ __launch_bounds__(256) void kprep(P p) {
  int b = blockIdx.x, t = threadIdx.x;
  if (b == 0) {                       // keytab[j][0..95] = MLP_key_ds(one_hot(j))
    __shared__ float tt[64]; __shared__ float st[2];
    for (int j = 0; j < 8; j++) {
      if (t < 52) tt[t] = p.kd_W1[j * 52 + t] + p.kd_b1[t];
      __syncthreads();
      if (t == 0) {
        float s = 0.f; for (int c = 0; c < 52; c++) s += tt[c];
        float mu = s / 52.f, q = 0.f;
        for (int c = 0; c < 52; c++) { float d = tt[c] - mu; q += d * d; }
        st[0] = mu; st[1] = rsqrtf(q / 52.f + EPS);
      }
      __syncthreads();
      if (t < 52) tt[t] = tanhf((tt[t] - st[0]) * st[1] * p.kd_g1[t] + p.kd_be1[t]);
      __syncthreads();
      if (t < 96) {
        float a = p.kd_b2[t];
        for (int c = 0; c < 52; c++) a += tt[c] * p.kd_W2[c * 96 + t];
        p.keytab[j * 96 + t] = a;
      }
      __syncthreads();
    }
  } else if (b == 1) {                // fold one_hot(n=8) into enc_main bias
    if (t < 132) p.em_b1p[t] = p.em_b1[t] + p.em_W1[136 * 132 + t];
  } else switch (b) {
    case 2:  convT(p.vd_W1, p.w1t_vd,  96, 96,  96,  96, t); break;
    case 3:  convT(p.vd_W2, p.w2t_vd,  96, 96,  96,  96, t); break;
    case 4:  convT(p.ed_W1, p.w1t_ed,  96, 96,  96,  96, t); break;
    case 5:  convT(p.ed_W2, p.w2t_ed,  96, 96,  96,  96, t); break;
    case 6:  convT(p.vm_W1, p.w1t_vm, 192,160, 192, 160, t); break;
    case 7:  convT(p.vm_W2, p.w2t_vm, 160,128, 160, 128, t); break;
    case 8:  convT(p.km_W1, p.w1t_km, 105,116, 128, 128, t); break;
    case 9:  convT(p.km_W2, p.w2t_km, 116,128, 128, 128, t); break;
    case 10: convT(p.em_W1, p.w1t_em, 128,132, 128, 144, t); break; // rows 128..136 folded/zero
    case 11: convT(p.em_W2, p.w2t_em, 132,128, 160, 128, t); break;
  }
}

// ---- rank magnitudes -----------------------------------------------------
__global__ __launch_bounds__(256) void kmag(P p) {
  int row = blockIdx.x * 256 + threadIdx.x;
  if (row >= NN) return;
  const float* xr = p.x + (size_t)row * 96;
  float a = p.rank_b[0];
  #pragma unroll 8
  for (int j = 0; j < 96; j++) a = fmaf(xr[j], p.rank_W[j], a);
  p.mag[row] = a;
}

// ---- stable 8-element sort per segment -> gather permutation ------------
__global__ __launch_bounds__(256) void ksort(P p) {
  int s = blockIdx.x * 256 + threadIdx.x;
  if (s >= NB) return;
  int b8 = s * 8;
  float m[8];
  #pragma unroll
  for (int i = 0; i < 8; i++) m[i] = p.mag[b8 + i];
  #pragma unroll
  for (int i = 0; i < 8; i++) {
    int r = 0;
    #pragma unroll
    for (int j = 0; j < 8; j++) r += (m[j] < m[i]) || (m[j] == m[i] && j < i);
    p.perm[b8 + r] = b8 + i;
  }
}

// ---- deepset pass: y2_ds[seg] = sum_seg val_ds(xs) * keytab[pos] --------
__global__ __launch_bounds__(128) void kds(P p) {
  __shared__ __align__(16) h16 S[64 * 104];          // xs f16, then Hh overlay
  const int tid = threadIdx.x, lane = tid & 31, wid = tid >> 5, m0 = wid * 16;
  const int base = blockIdx.x * 64;
  const int cl = lane & 15, hf = lane >> 4;
  // stage sorted rows (wave-private rows -> no barriers anywhere)
  for (int r = 0; r < 16; r++) {
    int row = m0 + r, src = p.perm[base + row];
    const float* xr = p.x + (size_t)src * 96;
    for (int c = lane; c < 96; c += 32) S[row * 104 + c] = (h16)xr[c];
  }
  // GEMM1: xs @ W1 (96x96)
  v8f acc[6];
  #pragma unroll
  for (int i = 0; i < 6; i++) acc[i] = vzero();
  #pragma unroll
  for (int k = 0; k < 3; k++) {
    v16h a = ldA(S, 104, m0, k * 32, lane);
    #pragma unroll
    for (int nt = 0; nt < 6; nt++) acc[nt] = wmmaf(a, ldB(p.w1t_vd, 96, nt * 16, k * 32, lane), acc[nt]);
  }
  // bias + LayerNorm(96) + tanh -> f16 back into S
  float s[8] = {0}, ss[8] = {0};
  #pragma unroll
  for (int nt = 0; nt < 6; nt++) {
    float bc = p.vd_b1[nt * 16 + cl];
    #pragma unroll
    for (int g = 0; g < 8; g++) { float v = acc[nt][g] + bc; acc[nt][g] = v; s[g] += v; ss[g] += v * v; }
  }
  #pragma unroll
  for (int m = 1; m < 16; m <<= 1)
    #pragma unroll
    for (int g = 0; g < 8; g++) { s[g] += __shfl_xor(s[g], m, 32); ss[g] += __shfl_xor(ss[g], m, 32); }
  float mu[8], rs[8];
  #pragma unroll
  for (int g = 0; g < 8; g++) { mu[g] = s[g] * (1.f/96.f); float var = ss[g]*(1.f/96.f) - mu[g]*mu[g]; rs[g] = rsqrtf(var + EPS); }
  #pragma unroll
  for (int nt = 0; nt < 6; nt++) {
    int col = nt * 16 + cl; float gam = p.vd_g1[col], bet = p.vd_be1[col];
    #pragma unroll
    for (int g = 0; g < 8; g++) {
      int row = m0 + g + 8 * hf;
      float v = (acc[nt][g] - mu[g]) * rs[g] * gam + bet;
      S[row * 104 + col] = (h16)tanhf(v);
    }
  }
  // GEMM2 + b2, * keytab[pos], segment sum (pos == g)
  v8f acc2[6];
  #pragma unroll
  for (int i = 0; i < 6; i++) acc2[i] = vzero();
  #pragma unroll
  for (int k = 0; k < 3; k++) {
    v16h a = ldA(S, 104, m0, k * 32, lane);
    #pragma unroll
    for (int nt = 0; nt < 6; nt++) acc2[nt] = wmmaf(a, ldB(p.w2t_vd, 96, nt * 16, k * 32, lane), acc2[nt]);
  }
  int seg = blockIdx.x * 8 + wid * 2 + hf;
  #pragma unroll
  for (int nt = 0; nt < 6; nt++) {
    int col = nt * 16 + cl; float b2 = p.vd_b2[col];
    float sum = 0.f;
    #pragma unroll
    for (int g = 0; g < 8; g++) sum += (acc2[nt][g] + b2) * p.keytab[g * 96 + col];
    p.y2ds[(size_t)seg * 96 + col] = sum;
  }
}

// ---- enc_ds: z_ds = MLP(96->96->96, tanh) over segments -----------------
__global__ __launch_bounds__(128) void keds(P p) {
  __shared__ __align__(16) h16 S[64 * 104];
  const int tid = threadIdx.x, lane = tid & 31, wid = tid >> 5, m0 = wid * 16;
  const int base = blockIdx.x * 64;
  const int cl = lane & 15, hf = lane >> 4;
  for (int r = 0; r < 16; r++) {
    int row = m0 + r; size_t grow = base + row;
    for (int c = lane; c < 96; c += 32) S[row * 104 + c] = (h16)p.y2ds[grow * 96 + c];
  }
  v8f acc[6];
  #pragma unroll
  for (int i = 0; i < 6; i++) acc[i] = vzero();
  #pragma unroll
  for (int k = 0; k < 3; k++) {
    v16h a = ldA(S, 104, m0, k * 32, lane);
    #pragma unroll
    for (int nt = 0; nt < 6; nt++) acc[nt] = wmmaf(a, ldB(p.w1t_ed, 96, nt * 16, k * 32, lane), acc[nt]);
  }
  #pragma unroll
  for (int nt = 0; nt < 6; nt++) {
    int col = nt * 16 + cl; float bc = p.ed_b1[col];
    #pragma unroll
    for (int g = 0; g < 8; g++) {
      int row = m0 + g + 8 * hf;
      S[row * 104 + col] = (h16)tanhf(acc[nt][g] + bc);
    }
  }
  v8f acc2[6];
  #pragma unroll
  for (int i = 0; i < 6; i++) acc2[i] = vzero();
  #pragma unroll
  for (int k = 0; k < 3; k++) {
    v16h a = ldA(S, 104, m0, k * 32, lane);
    #pragma unroll
    for (int nt = 0; nt < 6; nt++) acc2[nt] = wmmaf(a, ldB(p.w2t_ed, 96, nt * 16, k * 32, lane), acc2[nt]);
  }
  #pragma unroll
  for (int nt = 0; nt < 6; nt++) {
    int col = nt * 16 + cl; float b2 = p.ed_b2[col];
    #pragma unroll
    for (int g = 0; g < 8; g++) {
      size_t grow = base + m0 + g + 8 * hf;
      p.zdsh[grow * 96 + col] = (h16)(acc2[nt][g] + b2);
    }
  }
}

// ---- main pass: y2[seg] = sum_seg val_main(x_in) * key_main(pos_in) -----
__global__ __launch_bounds__(128) void kmain(P p) {
  __shared__ __align__(16) h16 SM[64 * 200 + 64 * 136];
  h16* Aval = SM;                 // [64][192] (also reused as Hh [64][<=160])
  h16* Akey = SM + 64 * 200;      // [64][128]
  const int tid = threadIdx.x, lane = tid & 31, wid = tid >> 5, m0 = wid * 16;
  const int base = blockIdx.x * 64;
  const int cl = lane & 15, hf = lane >> 4;
  // stage: Aval = [xs | z_ds];  Akey = [one_hot(pos) | z_ds | 0]
  for (int r = 0; r < 16; r++) {
    int row = m0 + r, grow = base + row, src = p.perm[grow], seg = grow >> 3;
    const float* xr = p.x + (size_t)src * 96;
    const h16* zr = p.zdsh + (size_t)seg * 96;
    for (int c = lane; c < 96; c += 32) {
      Aval[row * 200 + c] = (h16)xr[c];
      h16 zv = zr[c];
      Aval[row * 200 + 96 + c] = zv;
      Akey[row * 136 + 9 + c] = zv;
    }
    if (lane < 9) Akey[row * 136 + lane] = (lane == (grow & 7)) ? (h16)1.0f : (h16)0.0f;
    for (int c = 105 + lane; c < 128; c += 32) Akey[row * 136 + c] = (h16)0.0f;
  }
  // ---- val_main GEMM1: K=192 -> mid 160 ----
  v8f av[10];
  #pragma unroll
  for (int i = 0; i < 10; i++) av[i] = vzero();
  #pragma unroll
  for (int k = 0; k < 6; k++) {
    v16h a = ldA(Aval, 200, m0, k * 32, lane);
    #pragma unroll
    for (int nt = 0; nt < 10; nt++) av[nt] = wmmaf(a, ldB(p.w1t_vm, 192, nt * 16, k * 32, lane), av[nt]);
  }
  { // bias + LN(160) + tanh -> Hh (overlay Aval, wave-private rows)
    float s[8] = {0}, ss[8] = {0};
    #pragma unroll
    for (int nt = 0; nt < 10; nt++) {
      float bc = p.vm_b1[nt * 16 + cl];
      #pragma unroll
      for (int g = 0; g < 8; g++) { float v = av[nt][g] + bc; av[nt][g] = v; s[g] += v; ss[g] += v * v; }
    }
    #pragma unroll
    for (int m = 1; m < 16; m <<= 1)
      #pragma unroll
      for (int g = 0; g < 8; g++) { s[g] += __shfl_xor(s[g], m, 32); ss[g] += __shfl_xor(ss[g], m, 32); }
    #pragma unroll
    for (int nt = 0; nt < 10; nt++) {
      int col = nt * 16 + cl; float gam = p.vm_g1[col], bet = p.vm_be1[col];
      #pragma unroll
      for (int g = 0; g < 8; g++) {
        float mu = s[g] * (1.f/160.f), var = ss[g]*(1.f/160.f) - mu*mu, rs = rsqrtf(var + EPS);
        int row = m0 + g + 8 * hf;
        Aval[row * 200 + col] = (h16)tanhf((av[nt][g] - mu) * rs * gam + bet);
      }
    }
  }
  // ---- val_main GEMM2: K=160 -> 128 ----
  v8f av2[8];
  #pragma unroll
  for (int i = 0; i < 8; i++) av2[i] = vzero();
  #pragma unroll
  for (int k = 0; k < 5; k++) {
    v16h a = ldA(Aval, 200, m0, k * 32, lane);
    #pragma unroll
    for (int nt = 0; nt < 8; nt++) av2[nt] = wmmaf(a, ldB(p.w2t_vm, 160, nt * 16, k * 32, lane), av2[nt]);
  }
  // ---- key_main GEMM1: K=128 (pad of 105) -> mid 116 (pad 128) ----
  v8f ak[8];
  #pragma unroll
  for (int i = 0; i < 8; i++) ak[i] = vzero();
  #pragma unroll
  for (int k = 0; k < 4; k++) {
    v16h a = ldA(Akey, 136, m0, k * 32, lane);
    #pragma unroll
    for (int nt = 0; nt < 8; nt++) ak[nt] = wmmaf(a, ldB(p.w1t_km, 128, nt * 16, k * 32, lane), ak[nt]);
  }
  { // bias + LN over exactly 116 + tanh; zero the 12 padded mid cols
    float s[8] = {0}, ss[8] = {0};
    #pragma unroll
    for (int nt = 0; nt < 8; nt++) {
      int col = nt * 16 + cl; bool valid = col < 116;
      float bc = valid ? p.km_b1[col] : 0.f;
      #pragma unroll
      for (int g = 0; g < 8; g++) {
        float v = ak[nt][g] + bc; ak[nt][g] = v;
        if (valid) { s[g] += v; ss[g] += v * v; }
      }
    }
    #pragma unroll
    for (int m = 1; m < 16; m <<= 1)
      #pragma unroll
      for (int g = 0; g < 8; g++) { s[g] += __shfl_xor(s[g], m, 32); ss[g] += __shfl_xor(ss[g], m, 32); }
    #pragma unroll
    for (int nt = 0; nt < 8; nt++) {
      int col = nt * 16 + cl; bool valid = col < 116;
      float gam = valid ? p.km_g1[col] : 0.f, bet = valid ? p.km_be1[col] : 0.f;
      #pragma unroll
      for (int g = 0; g < 8; g++) {
        float mu = s[g] * (1.f/116.f), var = ss[g]*(1.f/116.f) - mu*mu, rs = rsqrtf(var + EPS);
        int row = m0 + g + 8 * hf;
        float t = valid ? tanhf((ak[nt][g] - mu) * rs * gam + bet) : 0.f;
        Aval[row * 200 + col] = (h16)t;
      }
    }
  }
  // ---- key_main GEMM2: K=128 (pad of 116) -> 128 ----
  v8f ak2[8];
  #pragma unroll
  for (int i = 0; i < 8; i++) ak2[i] = vzero();
  #pragma unroll
  for (int k = 0; k < 4; k++) {
    v16h a = ldA(Aval, 200, m0, k * 32, lane);
    #pragma unroll
    for (int nt = 0; nt < 8; nt++) ak2[nt] = wmmaf(a, ldB(p.w2t_km, 128, nt * 16, k * 32, lane), ak2[nt]);
  }
  // combine + segment sum (pos == g)
  int seg = blockIdx.x * 8 + wid * 2 + hf;
  #pragma unroll
  for (int nt = 0; nt < 8; nt++) {
    int col = nt * 16 + cl;
    float bv = p.vm_b2[col], bk = p.km_b2[col];
    float sum = 0.f;
    #pragma unroll
    for (int g = 0; g < 8; g++) sum += (av2[nt][g] + bv) * (ak2[nt][g] + bk);
    p.y2m[(size_t)seg * 128 + col] = sum;
  }
}

// ---- enc_main: z = MLP(concat(y2, one_hot(8))) = y2@W1[:128] + b1' ... ---
__global__ __launch_bounds__(128) void kem(P p) {
  __shared__ __align__(16) h16 SM[64 * 136 + 64 * 168];
  h16* A  = SM;              // [64][128]
  h16* Hh = SM + 64 * 136;   // [64][160] (K padded to 160, zeroed tail)
  const int tid = threadIdx.x, lane = tid & 31, wid = tid >> 5, m0 = wid * 16;
  const int base = blockIdx.x * 64;
  const int cl = lane & 15, hf = lane >> 4;
  for (int r = 0; r < 16; r++) {
    int row = m0 + r; size_t grow = base + row;
    for (int c = lane; c < 128; c += 32) A[row * 136 + c] = (h16)p.y2m[grow * 128 + c];
  }
  // GEMM1: K=128 -> mid 132 (pad 144)
  v8f acc[9];
  #pragma unroll
  for (int i = 0; i < 9; i++) acc[i] = vzero();
  #pragma unroll
  for (int k = 0; k < 4; k++) {
    v16h a = ldA(A, 136, m0, k * 32, lane);
    #pragma unroll
    for (int nt = 0; nt < 9; nt++) acc[nt] = wmmaf(a, ldB(p.w1t_em, 128, nt * 16, k * 32, lane), acc[nt]);
  }
  #pragma unroll
  for (int nt = 0; nt < 9; nt++) {
    int col = nt * 16 + cl; bool valid = col < 132;
    float bc = valid ? p.em_b1p[col] : 0.f;
    #pragma unroll
    for (int g = 0; g < 8; g++) {
      int row = m0 + g + 8 * hf;
      Hh[row * 168 + col] = valid ? (h16)tanhf(acc[nt][g] + bc) : (h16)0.0f;
    }
  }
  #pragma unroll
  for (int g = 0; g < 8; g++) {        // zero padded K cols 144..159
    int row = m0 + g + 8 * hf;
    Hh[row * 168 + 144 + cl] = (h16)0.0f;
  }
  // GEMM2: K=160 (pad of 132) -> 128
  v8f acc2[8];
  #pragma unroll
  for (int i = 0; i < 8; i++) acc2[i] = vzero();
  #pragma unroll
  for (int k = 0; k < 5; k++) {
    v16h a = ldA(Hh, 168, m0, k * 32, lane);
    #pragma unroll
    for (int nt = 0; nt < 8; nt++) acc2[nt] = wmmaf(a, ldB(p.w2t_em, 160, nt * 16, k * 32, lane), acc2[nt]);
  }
  #pragma unroll
  for (int nt = 0; nt < 8; nt++) {
    int col = nt * 16 + cl; float b2 = p.em_b2[col];
    #pragma unroll
    for (int g = 0; g < 8; g++) {
      int grow = base + m0 + g + 8 * hf;
      if (grow < NB) p.out[(size_t)grow * 128 + col] = acc2[nt][g] + b2;
    }
  }
}

extern "C" void kernel_launch(void* const* d_in, const int* in_sizes, int n_in,
                              void* d_out, int out_size, void* d_ws, size_t ws_size,
                              hipStream_t stream) {
  P p{};
  int i = 0;
  p.x = (const float*)d_in[i++];
  i++;  // batch (implicit: row/8)
  i++;  // n_batches
  p.kd_W1=(const float*)d_in[i++]; p.kd_b1=(const float*)d_in[i++]; p.kd_W2=(const float*)d_in[i++];
  p.kd_b2=(const float*)d_in[i++]; p.kd_g1=(const float*)d_in[i++]; p.kd_be1=(const float*)d_in[i++];
  p.vd_W1=(const float*)d_in[i++]; p.vd_b1=(const float*)d_in[i++]; p.vd_W2=(const float*)d_in[i++];
  p.vd_b2=(const float*)d_in[i++]; p.vd_g1=(const float*)d_in[i++]; p.vd_be1=(const float*)d_in[i++];
  p.ed_W1=(const float*)d_in[i++]; p.ed_b1=(const float*)d_in[i++]; p.ed_W2=(const float*)d_in[i++];
  p.ed_b2=(const float*)d_in[i++];
  p.km_W1=(const float*)d_in[i++]; p.km_b1=(const float*)d_in[i++]; p.km_W2=(const float*)d_in[i++];
  p.km_b2=(const float*)d_in[i++]; p.km_g1=(const float*)d_in[i++]; p.km_be1=(const float*)d_in[i++];
  p.vm_W1=(const float*)d_in[i++]; p.vm_b1=(const float*)d_in[i++]; p.vm_W2=(const float*)d_in[i++];
  p.vm_b2=(const float*)d_in[i++]; p.vm_g1=(const float*)d_in[i++]; p.vm_be1=(const float*)d_in[i++];
  p.em_W1=(const float*)d_in[i++]; p.em_b1=(const float*)d_in[i++]; p.em_W2=(const float*)d_in[i++];
  p.em_b2=(const float*)d_in[i++];
  p.rank_W=(const float*)d_in[i++]; p.rank_b=(const float*)d_in[i++];
  p.out = (float*)d_out;

  char* w = (char*)d_ws; size_t off = 0;
  auto alloc = [&](size_t bytes) -> void* {
    void* r = w + off; off = (off + bytes + 255) & ~(size_t)255; return r;
  };
  p.mag    = (float*)alloc((size_t)NN * 4);
  p.perm   = (int*)  alloc((size_t)NN * 4);
  p.y2ds   = (float*)alloc((size_t)NBp * 96 * 4);
  p.zdsh   = (h16*)  alloc((size_t)NBp * 96 * 2);
  p.y2m    = (float*)alloc((size_t)NBp * 128 * 4);
  p.keytab = (float*)alloc(8 * 96 * 4);
  p.em_b1p = (float*)alloc(160 * 4);
  p.w1t_vd = (h16*)alloc(96 * 96 * 2);
  p.w2t_vd = (h16*)alloc(96 * 96 * 2);
  p.w1t_ed = (h16*)alloc(96 * 96 * 2);
  p.w2t_ed = (h16*)alloc(96 * 96 * 2);
  p.w1t_vm = (h16*)alloc(160 * 192 * 2);
  p.w2t_vm = (h16*)alloc(128 * 160 * 2);
  p.w1t_km = (h16*)alloc(128 * 128 * 2);
  p.w2t_km = (h16*)alloc(128 * 128 * 2);
  p.w1t_em = (h16*)alloc(144 * 128 * 2);
  p.w2t_em = (h16*)alloc(128 * 160 * 2);

  kprep <<<12, 256, 0, stream>>>(p);
  kmag  <<<(NN + 255) / 256, 256, 0, stream>>>(p);
  ksort <<<(NB + 255) / 256, 256, 0, stream>>>(p);
  kds   <<<NN / 64, 128, 0, stream>>>(p);
  keds  <<<NBp / 64, 128, 0, stream>>>(p);
  kmain <<<NN / 64, 128, 0, stream>>>(p);
  kem   <<<NBp / 64, 128, 0, stream>>>(p);
}
